// QuantizedLinear_1984274890909
// MI455X (gfx1250) — matrix operand order repaired and checked by reading, not compiled, not run on hardware
//
#include <hip/hip_runtime.h>
#include <stdint.h>

typedef _Float16 v16h __attribute__((ext_vector_type(16)));
typedef _Float16 v8h  __attribute__((ext_vector_type(8)));
typedef _Float16 v4h  __attribute__((ext_vector_type(4)));
typedef float    v8f  __attribute__((ext_vector_type(8)));

#define K_DIM 4096
#define N_DIM 11008
#define M_DIM 8192
#define NGRP  32          // K / 128 quant groups

#define BM 256
#define BN 128
#define BK 32
#define BKP (BK + 8)      // padded LDS pitch (halves): 80B = 20 banks -> conflict-free frags
#define NK (K_DIM / BK)   // 128 k-steps

__launch_bounds__(256)
__global__ void qlin_wmma_f16(const float* __restrict__ X,
                              const int8_t* __restrict__ Q,
                              const float* __restrict__ S,
                              const float* __restrict__ Z,
                              float* __restrict__ C)
{
    __shared__ _Float16 As[2][BM][BKP];   // activations, f16 (40 KB)
    __shared__ _Float16 Bs[2][BN][BKP];   // dequantized weights, f16 (20 KB)

    const int tid  = threadIdx.x;
    const int lane = tid & 31;
    const int wave = tid >> 5;     // 0..7
    const int wm   = wave >> 1;    // 0..3 : 64 rows of C each
    const int wn   = wave & 1;     // 0..1 : 64 cols of C each
    const int l15  = lane & 15;
    const int hi   = lane >> 4;

    const int m0 = blockIdx.y * BM;
    const int n0 = blockIdx.x * BN;

    // ---- staging geometry ----
    // A: 256 rows x 32 f32 = 2048 float4 -> 8 per thread (32 rows/pass, 8 float4/row)
    const int arow = tid >> 3;          // 0..31
    const int acol = (tid & 7) << 2;    // 0,4,...,28
    // B: 128 rows x 32 bytes -> 1 uint4 (16 bytes) per thread
    const int brow = tid >> 1;          // 0..127
    const int bco  = (tid & 1) << 4;    // 0 or 16

    float4 aReg[8];
    uint4  bReg;
    float  sReg, zReg;

    const float*  Sp = S + (size_t)(n0 + brow) * NGRP;
    const float*  Zp = Z + (size_t)(n0 + brow) * NGRP;
    const int8_t* Qp = Q + (size_t)(n0 + brow) * K_DIM + bco;

    auto gload = [&](int ks) {
        const int k0 = ks * BK;
        #pragma unroll
        for (int p = 0; p < 8; ++p) {
            const float* src = X + (size_t)(m0 + p * 32 + arow) * K_DIM + (k0 + acol);
            aReg[p] = *(const float4*)src;
        }
        bReg = *(const uint4*)(Qp + k0);
        const int g = k0 >> 7;          // BK=32 always inside one quant group
        sReg = Sp[g];
        zReg = Zp[g];
        // warm L2 for step ks+2 (global_prefetch_b8)
        if (ks + 2 < NK) {
            __builtin_prefetch(X + (size_t)(m0 + arow) * K_DIM + (k0 + 2 * BK + acol), 0, 1);
            __builtin_prefetch(Qp + (k0 + 2 * BK), 0, 1);
        }
    };

    auto lstore = [&](int buf) {
        #pragma unroll
        for (int p = 0; p < 8; ++p) {
            v4h h;
            h[0] = (_Float16)aReg[p].x;
            h[1] = (_Float16)aReg[p].y;
            h[2] = (_Float16)aReg[p].z;
            h[3] = (_Float16)aReg[p].w;
            *(v4h*)(&As[buf][p * 32 + arow][acol]) = h;
        }
        const float s = sReg, z = zReg;
        uint32_t w[4] = {bReg.x, bReg.y, bReg.z, bReg.w};
        v8h h0, h1;
        #pragma unroll
        for (int d = 0; d < 2; ++d)
            #pragma unroll
            for (int b = 0; b < 4; ++b) {
                h0[d * 4 + b] = (_Float16)(((float)((w[d]     >> (8 * b)) & 0xFF) - z) * s);
                h1[d * 4 + b] = (_Float16)(((float)((w[d + 2] >> (8 * b)) & 0xFF) - z) * s);
            }
        _Float16* dst = &Bs[buf][brow][bco];
        *(v8h*)(dst)     = h0;
        *(v8h*)(dst + 8) = h1;
    };

    // A fragment (16x32 f16): halves 0..7 = K hi*8+0..7, halves 8..15 = K 16+hi*8+0..7
    auto loadA = [&](int buf, int i) -> v16h {
        const _Float16* row = &As[buf][wm * 64 + i * 16 + l15][0];
        v8h lo = *(const v8h*)(row + hi * 8);
        v8h hh = *(const v8h*)(row + 16 + hi * 8);
        return __builtin_shufflevector(lo, hh, 0,1,2,3,4,5,6,7,8,9,10,11,12,13,14,15);
    };
    // B fragment (32x16 f16): lane = N column, halves 0..7 = K hi*8+0..7, halves 8..15 = K 16+hi*8+0..7
    auto loadB = [&](int buf, int j) -> v16h {
        const _Float16* row = &Bs[buf][wn * 64 + j * 16 + l15][0];
        v8h lo = *(const v8h*)(row + hi * 8);
        v8h hh = *(const v8h*)(row + 16 + hi * 8);
        return __builtin_shufflevector(lo, hh, 0,1,2,3,4,5,6,7,8,9,10,11,12,13,14,15);
    };

    v8f acc[4][4] = {};

    gload(0);
    lstore(0);
    __syncthreads();

    int cur = 0;
    for (int ks = 0; ks < NK; ++ks) {
        const bool more = (ks + 1) < NK;
        if (more) gload(ks + 1);        // global loads overlap the WMMAs below

        v16h b0 = loadB(cur, 0);
        v16h b1 = loadB(cur, 1);
        v16h b2 = loadB(cur, 2);
        v16h b3 = loadB(cur, 3);

        // Software-pipeline A fragments: issue the ds_loads for tile i+1
        // before the WMMA quad for tile i, so dscnt waits hide under WMMAs.
        v16h a = loadA(cur, 0);
        #pragma unroll
        for (int i = 0; i < 4; ++i) {
            v16h an;
            if (i < 3) an = loadA(cur, i + 1);
            acc[i][0] = __builtin_amdgcn_wmma_f32_16x16x32_f16(false, a, false, b0, (short)0, acc[i][0], false, false);
            acc[i][1] = __builtin_amdgcn_wmma_f32_16x16x32_f16(false, a, false, b1, (short)0, acc[i][1], false, false);
            acc[i][2] = __builtin_amdgcn_wmma_f32_16x16x32_f16(false, a, false, b2, (short)0, acc[i][2], false, false);
            acc[i][3] = __builtin_amdgcn_wmma_f32_16x16x32_f16(false, a, false, b3, (short)0, acc[i][3], false, false);
            a = an;
        }

        if (more) lstore(cur ^ 1);      // fill the other buffer (readers done last iter)
        __syncthreads();
        cur ^= 1;
    }

    // Epilogue: C/D layout -> VGPR r is row r + 8*hi of the 16x16 tile, lane%16 is the column
    float* Cb = C + (size_t)(m0 + wm * 64) * N_DIM + (n0 + wn * 64);
    #pragma unroll
    for (int i = 0; i < 4; ++i)
        #pragma unroll
        for (int j = 0; j < 4; ++j)
            #pragma unroll
            for (int r = 0; r < 8; ++r) {
                const int m = i * 16 + hi * 8 + r;
                const int n = j * 16 + l15;
                Cb[(size_t)m * N_DIM + n] = acc[i][j][r];
            }
}

extern "C" void kernel_launch(void* const* d_in, const int* in_sizes, int n_in,
                              void* d_out, int out_size, void* d_ws, size_t ws_size,
                              hipStream_t stream) {
    const float*  x  = (const float*)d_in[0];
    const int8_t* qw = (const int8_t*)d_in[1];
    const float*  sc = (const float*)d_in[2];
    const float*  zp = (const float*)d_in[3];
    float* out = (float*)d_out;

    dim3 grid(N_DIM / BN, M_DIM / BM);   // 86 x 32
    qlin_wmma_f16<<<grid, 256, 0, stream>>>(x, qw, sc, zp, out);
}
